// RAFTStereo_73358041416351
// MI455X (gfx1250) — matrix operand order repaired and checked by reading, not compiled, not run on hardware
//
#include <hip/hip_runtime.h>

// ---------------------------------------------------------------------------
// Types for gfx1250 WMMA (wave32). A/B: v16h (16 halves / lane), C/D: v8f.
// ---------------------------------------------------------------------------
typedef _Float16 h8   __attribute__((ext_vector_type(8)));
typedef _Float16 v16h __attribute__((ext_vector_type(16)));
typedef float    v8f  __attribute__((ext_vector_type(8)));

static __device__ __forceinline__ v16h zero_v16h() {
  v16h z;
#pragma unroll
  for (int i = 0; i < 16; ++i) z[i] = (_Float16)0.f;
  return z;
}

static __device__ __forceinline__ v16h combine16(h8 lo, h8 hi) {
  return __builtin_shufflevector(lo, hi, 0, 1, 2, 3, 4, 5, 6, 7, 8, 9, 10, 11, 12, 13, 14, 15);
}

static __device__ __forceinline__ float apply_act(float v, int act) {
  if (act == 1) return v > 0.f ? v : 0.f;           // relu
  if (act == 2) return 1.f / (1.f + __expf(-v));    // sigmoid
  if (act == 3) return tanhf(v);                    // tanh
  return v;
}

// ---------------------------------------------------------------------------
// Pack raw f32 weights [Cout, Cin, KH, KW] into WMMA B fragments in this file's contiguous k order:
// wpk[coTile][tap][kcTile][lane][16 halves], lane: N = lane&15, K = e + 16*(lane>>4)
// Zero-pads Cin -> CinPad and Cout -> CoutPad.
// ---------------------------------------------------------------------------
__global__ void k_pack_w(const float* __restrict__ w, _Float16* __restrict__ wpk,
                         int Cout, int Cin, int CinPad, int KH, int KW, int CoutPad) {
  const int kcT = CinPad >> 5, taps = KH * KW, coT = CoutPad >> 4;
  const int total = coT * taps * kcT * 32;
  int idx = blockIdx.x * blockDim.x + threadIdx.x;
  if (idx >= total) return;
  const int lane = idx & 31;
  int t = idx >> 5;
  const int kc = t % kcT; t /= kcT;
  const int tap = t % taps; t /= taps;
  const int ct = t;
  const int co = ct * 16 + (lane & 15);
  const int kbase = kc * 32 + 16 * (lane >> 4);
  _Float16* dst = wpk + (size_t)idx * 16;
#pragma unroll 4
  for (int e = 0; e < 16; ++e) {
    const int c = kbase + e;
    float v = (co < Cout && c < Cin) ? w[((size_t)co * Cin + c) * taps + tap] : 0.f;
    dst[e] = (_Float16)v;
  }
}

// NCHW f32 -> NHWC f16 (used for fmap1/fmap2 transpose-cast)
__global__ void k_to_nhwc_f16(const float* __restrict__ src, _Float16* __restrict__ dst,
                              int C, int HW, int total) {
  int idx = blockIdx.x * blockDim.x + threadIdx.x;
  if (idx >= total) return;
  const int c = idx % C;
  const int p = idx / C;
  const int n = p / HW, hw = p - n * HW;
  dst[idx] = (_Float16)src[((size_t)n * C + c) * HW + hw];
}

// ---------------------------------------------------------------------------
// Correlation: corr[b,h,i,j] = (1/16) * sum_c f1[b,h,i,c] * f2[b,h,j,c]
// 2x2 register-blocked: one wave computes a 32x32 (i,j) tile, K=256.
// ---------------------------------------------------------------------------
__global__ void k_corr_wmma(const _Float16* __restrict__ f1, const _Float16* __restrict__ f2,
                            float* __restrict__ corr, int BH, int W, int C) {
  const int lane = threadIdx.x & 31;
  const int gwave = (blockIdx.x * blockDim.x + threadIdx.x) >> 5;
  const int WT2 = W >> 5;                    // 32-wide tiles (5)
  const int tilesPerSlab = WT2 * WT2;        // 25
  const int slab = gwave / tilesPerSlab;
  if (slab >= BH) return;
  const int t = gwave - slab * tilesPerSlab;
  const int it = t / WT2, jt = t - it * WT2;
  const int hi = lane >> 4;
  const int i0 = it * 32 + (lane & 15);
  const int j0 = jt * 32 + (lane & 15);
  const _Float16* a0p = f1 + ((size_t)slab * W + i0) * C + hi * 16;
  const _Float16* a1p = a0p + (size_t)16 * C;
  const _Float16* b0p = f2 + ((size_t)slab * W + j0) * C + hi * 16;
  const _Float16* b1p = b0p + (size_t)16 * C;
  v8f acc00 = {0.f, 0.f, 0.f, 0.f, 0.f, 0.f, 0.f, 0.f};
  v8f acc01 = acc00, acc10 = acc00, acc11 = acc00;
  const int kcT = C >> 5;
  for (int kc = 0; kc < kcT; ++kc) {
    v16h a0 = combine16(*(const h8*)(a0p + kc * 32), *(const h8*)(a0p + kc * 32 + 8));
    v16h a1 = combine16(*(const h8*)(a1p + kc * 32), *(const h8*)(a1p + kc * 32 + 8));
    v16h b0 = combine16(*(const h8*)(b0p + kc * 32), *(const h8*)(b0p + kc * 32 + 8));
    v16h b1 = combine16(*(const h8*)(b1p + kc * 32), *(const h8*)(b1p + kc * 32 + 8));
    acc00 = __builtin_amdgcn_wmma_f32_16x16x32_f16(false, a0, false, b0, (short)0, acc00, false, false);
    acc01 = __builtin_amdgcn_wmma_f32_16x16x32_f16(false, a0, false, b1, (short)0, acc01, false, false);
    acc10 = __builtin_amdgcn_wmma_f32_16x16x32_f16(false, a1, false, b0, (short)0, acc10, false, false);
    acc11 = __builtin_amdgcn_wmma_f32_16x16x32_f16(false, a1, false, b1, (short)0, acc11, false, false);
  }
  const float s = 0.0625f;
#pragma unroll
  for (int v = 0; v < 8; ++v) {
    const int r0 = it * 32 + v + hi * 8;
    const int r1 = r0 + 16;
    const int c0 = jt * 32 + (lane & 15);
    float* row0 = corr + ((size_t)slab * W + r0) * W;
    float* row1 = corr + ((size_t)slab * W + r1) * W;
    row0[c0] = acc00[v] * s;
    row0[c0 + 16] = acc01[v] * s;
    row1[c0] = acc10[v] * s;
    row1[c0 + 16] = acc11[v] * s;
  }
}

// Pyramid pooling along last dim: out[r,j] = 0.5*(in[r,2j] + in[r,2j+1])
__global__ void k_pool(const float* __restrict__ in, float* __restrict__ out,
                       int Wl2, int total) {
  int idx = blockIdx.x * blockDim.x + threadIdx.x;
  if (idx >= total) return;
  const int j = idx % Wl2;
  const size_t r = idx / Wl2;
  const float* ip = in + r * (size_t)(Wl2 * 2) + 2 * j;
  out[idx] = 0.5f * (ip[0] + ip[1]);
}

// Correlation lookup -> NHWC f16 [NP, 64] (channels 36..63 zeroed for Cin pad)
__global__ void k_lookup(const float* __restrict__ p0, const float* __restrict__ p1,
                         const float* __restrict__ p2, const float* __restrict__ p3,
                         const float* __restrict__ coords, _Float16* __restrict__ corr36,
                         int W, int NP) {
  int idx = blockIdx.x * blockDim.x + threadIdx.x;
  if (idx >= NP * 64) return;
  const int c = idx & 63;
  const int p = idx >> 6;
  if (c >= 36) { corr36[idx] = (_Float16)0.f; return; }
  const int l = c / 9, t = c - 9 * l;
  const float* lv = (l == 0) ? p0 : (l == 1) ? p1 : (l == 2) ? p2 : p3;
  const int Wl = W >> l;
  const float x = coords[p] / (float)(1 << l) + (float)(t - 4);
  const float x0 = floorf(x);
  const float f = x - x0;
  const int xi = (int)x0;
  const float* row = lv + (size_t)p * Wl;
  const float g0 = (xi >= 0 && xi < Wl) ? row[xi] : 0.f;
  const float g1 = (xi + 1 >= 0 && xi + 1 < Wl) ? row[xi + 1] : 0.f;
  corr36[idx] = (_Float16)(g0 * (1.f - f) + g1 * f);
}

// net = tanh(cnet[:, :128]); inp = relu(cnet[:, 128:]); fill cat384 channels.
__global__ void k_init_state(const float* __restrict__ cnet, float* __restrict__ netf,
                             _Float16* __restrict__ cat384, int HW, int NP) {
  int idx = blockIdx.x * blockDim.x + threadIdx.x;
  if (idx >= NP * 128) return;
  const int c = idx & 127;
  const int p = idx >> 7;
  const int n = p / HW, hw = p - n * HW;
  const float nv = tanhf(cnet[((size_t)n * 256 + c) * HW + hw]);
  float iv = cnet[((size_t)n * 256 + 128 + c) * HW + hw];
  iv = iv > 0.f ? iv : 0.f;
  netf[idx] = nv;
  cat384[(size_t)p * 384 + c] = (_Float16)nv;
  cat384[(size_t)p * 384 + 256 + c] = (_Float16)iv;
}

__global__ void k_init_coords(float* __restrict__ coords, int W, int NP) {
  int p = blockIdx.x * blockDim.x + threadIdx.x;
  if (p >= NP) return;
  coords[p] = (float)(p % W);
}

// disp = coords - org; feed f32 planar (convf1) and f16 into cat384 ch 255.
__global__ void k_disp_prep(const float* __restrict__ coords, float* __restrict__ dispP,
                            _Float16* __restrict__ cat384, int W, int NP) {
  int p = blockIdx.x * blockDim.x + threadIdx.x;
  if (p >= NP) return;
  const float d = coords[p] - (float)(p % W);
  dispP[p] = d;
  cat384[(size_t)p * 384 + 255] = (_Float16)d;
}

// convf1: 7x7 conv, Cin=1, Cout=64, relu (K=49 -> direct VALU is fine).
__global__ void k_conv7(const float* __restrict__ disp, const float* __restrict__ w,
                        const float* __restrict__ bias, _Float16* __restrict__ out,
                        int H, int W, int NP) {
  int idx = blockIdx.x * blockDim.x + threadIdx.x;
  if (idx >= NP * 64) return;
  const int co = idx & 63;
  const int p = idx >> 6;
  const int x = p % W;
  const int y = (p / W) % H;
  float s = bias[co];
  const float* wc = w + co * 49;
  for (int ky = 0; ky < 7; ++ky) {
    const int yy = y + ky - 3;
    if (yy < 0 || yy >= H) continue;
    for (int kx = 0; kx < 7; ++kx) {
      const int xx = x + kx - 3;
      if (xx < 0 || xx >= W) continue;
      s += wc[ky * 7 + kx] * disp[p + (ky - 3) * W + (kx - 3)];
    }
  }
  out[idx] = (_Float16)(s > 0.f ? s : 0.f);
}

// ---------------------------------------------------------------------------
// Implicit-GEMM convolution via WMMA f16 (f32 accumulate), 2x2 register tiles.
//   grid = (pixelPairs/8, coutPairs); block = 256 threads = 8 waves.
//   All 8 waves of a block share the same pair of cout tiles, so the packed
//   weight fragments for each tap are staged once per block into LDS with
//   GLOBAL_LOAD_ASYNC_TO_LDS_B128 (ASYNCcnt), double-buffered across taps,
//   and consumed via ds_load. Activations (A) stream from global as b128.
//   Requires: NP % 32 == 0 and pixelPairs % 8 == 0 (29440/32 = 920 = 115*8).
// ---------------------------------------------------------------------------
__global__ void k_conv_wmma(const _Float16* __restrict__ in, int inCtot,
                            const _Float16* __restrict__ wpk, const float* __restrict__ bias,
                            _Float16* __restrict__ outh, int outhCtot, int outhCoff,
                            float* __restrict__ outf, int outfCtot, int outfCoff,
                            int NP, int H, int W, int KH, int KW, int pad,
                            int kcT, int Cout, int act, float scale) {
  extern __shared__ _Float16 smem[];   // [2 bufs][2 ct][kcT][32 lanes][16 halves]

  const int lane = threadIdx.x & 31;
  const int wib = threadIdx.x >> 5;          // wave in block: 0..7
  const int pp = blockIdx.x * 8 + wib;       // pixel-pair index
  const int cp = blockIdx.y;                 // cout-pair index
  const int taps = KH * KW;
  const int hi = lane >> 4;
  const int ctHalves = kcT * 512;            // halves per (ct, tap) chunk

  // A-side pixels for this lane (rows of the two pixel tiles)
  const int pa0 = pp * 32 + (lane & 15);
  const int pa1 = pa0 + 16;
  const int xa0 = pa0 % W, ya0 = (pa0 / W) % H;
  const int xa1 = pa1 % W, ya1 = (pa1 / W) % H;

  // Async staging of one tap's weights (both cout tiles) into LDS buffer `buf`
  auto stage = [&](int tap, int buf) {
    const int chunks = kcT * 128;            // 2 ct * kcT * 64 chunks of 16B
    for (int c = threadIdx.x; c < chunks; c += 256) {
      const int ctl = c / (kcT * 64);
      const int rem = c - ctl * (kcT * 64);
      const _Float16* src = wpk + ((size_t)(cp * 2 + ctl) * taps + tap) * ctHalves +
                            (size_t)rem * 8;
      unsigned dst = (unsigned)(size_t)&smem[(size_t)(buf * 2 + ctl) * ctHalves + rem * 8];
      asm volatile("global_load_async_to_lds_b128 %0, %1, off"
                   :: "v"(dst), "v"(src) : "memory");
    }
  };

  v8f acc00 = {0.f, 0.f, 0.f, 0.f, 0.f, 0.f, 0.f, 0.f};
  v8f acc01 = acc00, acc10 = acc00, acc11 = acc00;

  stage(0, 0);
  for (int tap = 0; tap < taps; ++tap) {
    asm volatile("s_wait_asynccnt 0x0" ::: "memory");
    __syncthreads();                          // buf(tap) ready for whole block
    if (tap + 1 < taps) stage(tap + 1, (tap + 1) & 1);

    const int buf = tap & 1;
    const int ky = tap / KW - pad;
    const int kx = tap % KW - pad;
    const int yy0 = ya0 + ky, xx0 = xa0 + kx;
    const int yy1 = ya1 + ky, xx1 = xa1 + kx;
    const bool inb0 = (yy0 >= 0) & (yy0 < H) & (xx0 >= 0) & (xx0 < W);
    const bool inb1 = (yy1 >= 0) & (yy1 < H) & (xx1 >= 0) & (xx1 < W);
    const _Float16* a0base = in + (long)(pa0 + ky * W + kx) * inCtot + hi * 16;
    const _Float16* a1base = in + (long)(pa1 + ky * W + kx) * inCtot + hi * 16;
    const _Float16* bs0 = &smem[(size_t)(buf * 2) * ctHalves + lane * 16];
    const _Float16* bs1 = bs0 + ctHalves;

    for (int kc = 0; kc < kcT; ++kc) {
      v16h a0 = inb0 ? combine16(*(const h8*)(a0base + kc * 32),
                                 *(const h8*)(a0base + kc * 32 + 8))
                     : zero_v16h();
      v16h a1 = inb1 ? combine16(*(const h8*)(a1base + kc * 32),
                                 *(const h8*)(a1base + kc * 32 + 8))
                     : zero_v16h();
      v16h b0 = combine16(*(const h8*)(bs0 + kc * 512), *(const h8*)(bs0 + kc * 512 + 8));
      v16h b1 = combine16(*(const h8*)(bs1 + kc * 512), *(const h8*)(bs1 + kc * 512 + 8));
      __builtin_prefetch(a0base + (kc + 1) * 32, 0, 0);   // global_prefetch_b8
      acc00 = __builtin_amdgcn_wmma_f32_16x16x32_f16(false, a0, false, b0, (short)0, acc00, false, false);
      acc01 = __builtin_amdgcn_wmma_f32_16x16x32_f16(false, a0, false, b1, (short)0, acc01, false, false);
      acc10 = __builtin_amdgcn_wmma_f32_16x16x32_f16(false, a1, false, b0, (short)0, acc10, false, false);
      acc11 = __builtin_amdgcn_wmma_f32_16x16x32_f16(false, a1, false, b1, (short)0, acc11, false, false);
    }
    __syncthreads();                          // all waves done with buf before re-stage
  }

  // Epilogue: act(scale*(acc+bias)); optional f16 / f32 NHWC outputs.
  const int coBase = cp * 32 + (lane & 15);
#pragma unroll
  for (int cj = 0; cj < 2; ++cj) {
    const int co = coBase + cj * 16;
    if (co >= Cout) continue;
    const float bv = bias ? bias[co] : 0.f;
#pragma unroll
    for (int ti = 0; ti < 2; ++ti) {
      const v8f& a = (cj == 0) ? (ti == 0 ? acc00 : acc10) : (ti == 0 ? acc01 : acc11);
#pragma unroll
      for (int v = 0; v < 8; ++v) {
        const int pe = pp * 32 + ti * 16 + v + hi * 8;
        const float val = apply_act(scale * (a[v] + bv), act);
        if (outh) outh[(size_t)pe * outhCtot + outhCoff + co] = (_Float16)val;
        if (outf) outf[(size_t)pe * outfCtot + outfCoff + co] = val;
      }
    }
  }
}

// cat384[:, 0:128] = f16(r * net)
__global__ void k_rnet(const _Float16* __restrict__ rq, const float* __restrict__ netf,
                       _Float16* __restrict__ cat384, int NP) {
  int idx = blockIdx.x * blockDim.x + threadIdx.x;
  if (idx >= NP * 128) return;
  const int c = idx & 127;
  const int p = idx >> 7;
  cat384[(size_t)p * 384 + c] = (_Float16)((float)rq[idx] * netf[idx]);
}

// net = (1-z)*net + z*q; refresh f16 copy in cat384[:, 0:128]
__global__ void k_gru(const _Float16* __restrict__ zb, const _Float16* __restrict__ qb,
                      float* __restrict__ netf, _Float16* __restrict__ cat384, int NP) {
  int idx = blockIdx.x * blockDim.x + threadIdx.x;
  if (idx >= NP * 128) return;
  const int c = idx & 127;
  const int p = idx >> 7;
  const float z = (float)zb[idx];
  const float q = (float)qb[idx];
  float n = netf[idx];
  n = (1.f - z) * n + z * q;
  netf[idx] = n;
  cat384[(size_t)p * 384 + c] = (_Float16)n;
}

__global__ void k_coords_upd(float* __restrict__ coords, const float* __restrict__ delta,
                             float* __restrict__ disp2, int W, int NP) {
  int p = blockIdx.x * blockDim.x + threadIdx.x;
  if (p >= NP) return;
  const float c = coords[p] + delta[p];
  coords[p] = c;
  disp2[p] = c - (float)(p % W);
}

// Convex upsample: softmax over 9 mask weights x 3x3 neighborhood of 8*disp.
__global__ void k_upsample(const float* __restrict__ mask, const float* __restrict__ disp2,
                           float* __restrict__ out, int H, int W, int B) {
  const int OW = W * 8, OH = H * 8;
  int idx = blockIdx.x * blockDim.x + threadIdx.x;
  if (idx >= B * OH * OW) return;
  const int ox = idx % OW;
  const int r = idx / OW;
  const int oy = r % OH;
  const int b = r / OH;
  const int x = ox >> 3, j = ox & 7, y = oy >> 3, i = oy & 7;
  const size_t p = (size_t)(b * H + y) * W + x;
  float m[9];
  float mx = -1e30f;
#pragma unroll
  for (int k = 0; k < 9; ++k) {
    m[k] = mask[p * 576 + k * 64 + i * 8 + j];
    mx = fmaxf(mx, m[k]);
  }
  float se = 0.f, acc = 0.f;
#pragma unroll
  for (int k = 0; k < 9; ++k) {
    const float e = __expf(m[k] - mx);
    se += e;
    const int yy = y + k / 3 - 1, xx = x + k % 3 - 1;
    const float v = (yy >= 0 && yy < H && xx >= 0 && xx < W)
                        ? 8.f * disp2[(size_t)(b * H + yy) * W + xx]
                        : 0.f;
    acc += e * v;
  }
  out[idx] = acc / se;
}

// ---------------------------------------------------------------------------
// Host orchestration
// ---------------------------------------------------------------------------
extern "C" void kernel_launch(void* const* d_in, const int* in_sizes, int n_in,
                              void* d_out, int out_size, void* d_ws, size_t ws_size,
                              hipStream_t stream) {
  (void)in_sizes; (void)n_in; (void)ws_size;
  const int B = 2, H = 92, W = 160;
  const int HW = H * W, NP = B * HW, BH = B * H;
  const int OH = 8 * H, OW = 8 * W;
  const int iters = out_size / (B * OH * OW);   // = 12

  const float* fmap1 = (const float*)d_in[0];
  const float* fmap2 = (const float*)d_in[1];
  const float* cnet1 = (const float*)d_in[2];
  const float* P[24];
  for (int i = 0; i < 24; ++i) P[i] = (const float*)d_in[3 + i];
  const float *convc1_w = P[0], *convc1_b = P[1], *convc2_w = P[2], *convc2_b = P[3];
  const float *convf1_w = P[4], *convf1_b = P[5], *convf2_w = P[6], *convf2_b = P[7];
  const float *conv_w = P[8], *conv_b = P[9];
  const float *convz_w = P[10], *convz_b = P[11], *convr_w = P[12], *convr_b = P[13];
  const float *convq_w = P[14], *convq_b = P[15];
  const float *fh1_w = P[16], *fh1_b = P[17], *fh2_w = P[18], *fh2_b = P[19];
  const float *mh1_w = P[20], *mh1_b = P[21], *mh2_w = P[22], *mh2_b = P[23];

  char* wsp = (char*)d_ws;
  auto alloc = [&](size_t bytes) -> void* {
    void* p = (void*)wsp;
    wsp += (bytes + 255) & ~(size_t)255;
    return p;
  };

  _Float16* f1t   = (_Float16*)alloc((size_t)NP * 256 * 2);
  _Float16* f2t   = (_Float16*)alloc((size_t)NP * 256 * 2);
  float* pyr0     = (float*)alloc((size_t)BH * W * 160 * 4);
  float* pyr1     = (float*)alloc((size_t)BH * W * 80 * 4);
  float* pyr2     = (float*)alloc((size_t)BH * W * 40 * 4);
  float* pyr3     = (float*)alloc((size_t)BH * W * 20 * 4);
  _Float16* corr36 = (_Float16*)alloc((size_t)NP * 64 * 2);
  _Float16* cor1   = (_Float16*)alloc((size_t)NP * 64 * 2);
  _Float16* flo1   = (_Float16*)alloc((size_t)NP * 64 * 2);
  _Float16* corflo = (_Float16*)alloc((size_t)NP * 128 * 2);
  _Float16* cat384 = (_Float16*)alloc((size_t)NP * 384 * 2);
  _Float16* zb     = (_Float16*)alloc((size_t)NP * 128 * 2);
  _Float16* rqb    = (_Float16*)alloc((size_t)NP * 128 * 2);
  float* netf      = (float*)alloc((size_t)NP * 128 * 4);
  _Float16* t256   = (_Float16*)alloc((size_t)NP * 256 * 2);
  float* delta     = (float*)alloc((size_t)NP * 4);
  float* maskb     = (float*)alloc((size_t)NP * 576 * 4);
  float* dispP     = (float*)alloc((size_t)NP * 4);
  float* disp2     = (float*)alloc((size_t)NP * 4);
  float* coords    = (float*)alloc((size_t)NP * 4);

  auto wpk_alloc = [&](int CoutPad, int taps, int CinPad) -> _Float16* {
    return (_Float16*)alloc((size_t)(CoutPad / 16) * taps * (CinPad / 32) * 512 * 2);
  };
  _Float16* w_c1 = wpk_alloc(64, 1, 64);
  _Float16* w_c2 = wpk_alloc(64, 9, 64);
  _Float16* w_f2 = wpk_alloc(64, 9, 64);
  _Float16* w_cv = wpk_alloc(128, 9, 128);
  _Float16* w_z  = wpk_alloc(128, 9, 384);
  _Float16* w_r  = wpk_alloc(128, 9, 384);
  _Float16* w_q  = wpk_alloc(128, 9, 384);
  _Float16* w_f1h = wpk_alloc(256, 9, 128);
  _Float16* w_f2h = wpk_alloc(32, 9, 256);
  _Float16* w_m1 = wpk_alloc(256, 9, 128);
  _Float16* w_m2 = wpk_alloc(576, 1, 256);

  const int TB = 256;
  auto grid = [&](long n) { return dim3((unsigned)((n + TB - 1) / TB)); };

  auto pack = [&](const float* w, _Float16* wpk, int Cout, int Cin, int CinPad,
                  int KH, int KW, int CoutPad) {
    long tot = (long)(CoutPad / 16) * KH * KW * (CinPad / 32) * 32;
    k_pack_w<<<grid(tot), TB, 0, stream>>>(w, wpk, Cout, Cin, CinPad, KH, KW, CoutPad);
  };
  pack(convc1_w, w_c1, 64, 36, 64, 1, 1, 64);
  pack(convc2_w, w_c2, 64, 64, 64, 3, 3, 64);
  pack(convf2_w, w_f2, 64, 64, 64, 3, 3, 64);
  pack(conv_w,   w_cv, 127, 128, 128, 3, 3, 128);
  pack(convz_w,  w_z, 128, 384, 384, 3, 3, 128);
  pack(convr_w,  w_r, 128, 384, 384, 3, 3, 128);
  pack(convq_w,  w_q, 128, 384, 384, 3, 3, 128);
  pack(fh1_w,    w_f1h, 256, 128, 128, 3, 3, 256);
  pack(fh2_w,    w_f2h, 1, 256, 256, 3, 3, 32);
  pack(mh1_w,    w_m1, 256, 128, 128, 3, 3, 256);
  pack(mh2_w,    w_m2, 576, 256, 256, 1, 1, 576);

  k_to_nhwc_f16<<<grid((long)NP * 256), TB, 0, stream>>>(fmap1, f1t, 256, HW, NP * 256);
  k_to_nhwc_f16<<<grid((long)NP * 256), TB, 0, stream>>>(fmap2, f2t, 256, HW, NP * 256);
  k_init_state<<<grid((long)NP * 128), TB, 0, stream>>>(cnet1, netf, cat384, HW, NP);
  k_init_coords<<<grid(NP), TB, 0, stream>>>(coords, W, NP);

  {
    long waves = (long)BH * (W / 32) * (W / 32);
    k_corr_wmma<<<grid(waves * 32), TB, 0, stream>>>(f1t, f2t, pyr0, BH, W, 256);
    long rows = (long)BH * W;
    k_pool<<<grid(rows * 80), TB, 0, stream>>>(pyr0, pyr1, 80, (int)(rows * 80));
    k_pool<<<grid(rows * 40), TB, 0, stream>>>(pyr1, pyr2, 40, (int)(rows * 40));
    k_pool<<<grid(rows * 20), TB, 0, stream>>>(pyr2, pyr3, 20, (int)(rows * 20));
  }

  // conv launcher: grid = (pixelPairs/8, coutPairs), dynamic LDS = 2*2*kcT KB
  auto conv = [&](const _Float16* in, int inCtot, const _Float16* wpk, const float* bias,
                  _Float16* outh, int outhCtot, int outhCoff,
                  float* outf, int outfCtot, int outfCoff,
                  int KH, int KW, int pad, int kcT, int Cout, int coT, int act, float scale) {
    dim3 g((unsigned)(NP / 32 / 8), (unsigned)(coT / 2));
    size_t shb = (size_t)kcT * 4096;          // 2 bufs * 2 ct * kcT * 1KB
    k_conv_wmma<<<g, TB, shb, stream>>>(in, inCtot, wpk, bias, outh, outhCtot, outhCoff,
                                        outf, outfCtot, outfCoff, NP, H, W, KH, KW, pad,
                                        kcT, Cout, act, scale);
  };

  for (int it = 0; it < iters; ++it) {
    k_lookup<<<grid((long)NP * 64), TB, 0, stream>>>(pyr0, pyr1, pyr2, pyr3, coords,
                                                     corr36, W, NP);
    k_disp_prep<<<grid(NP), TB, 0, stream>>>(coords, dispP, cat384, W, NP);

    // motion encoder
    conv(corr36, 64, w_c1, convc1_b, cor1, 64, 0, nullptr, 0, 0, 1, 1, 0, 2, 64, 4, 1, 1.f);
    conv(cor1, 64, w_c2, convc2_b, corflo, 128, 0, nullptr, 0, 0, 3, 3, 1, 2, 64, 4, 1, 1.f);
    k_conv7<<<grid((long)NP * 64), TB, 0, stream>>>(dispP, convf1_w, convf1_b, flo1, H, W, NP);
    conv(flo1, 64, w_f2, convf2_b, corflo, 128, 64, nullptr, 0, 0, 3, 3, 1, 2, 64, 4, 1, 1.f);
    conv(corflo, 128, w_cv, conv_b, cat384, 384, 128, nullptr, 0, 0, 3, 3, 1, 4, 127, 8, 1, 1.f);

    // ConvGRU
    conv(cat384, 384, w_z, convz_b, zb, 128, 0, nullptr, 0, 0, 3, 3, 1, 12, 128, 8, 2, 1.f);
    conv(cat384, 384, w_r, convr_b, rqb, 128, 0, nullptr, 0, 0, 3, 3, 1, 12, 128, 8, 2, 1.f);
    k_rnet<<<grid((long)NP * 128), TB, 0, stream>>>(rqb, netf, cat384, NP);
    conv(cat384, 384, w_q, convq_b, rqb, 128, 0, nullptr, 0, 0, 3, 3, 1, 12, 128, 8, 3, 1.f);
    k_gru<<<grid((long)NP * 128), TB, 0, stream>>>(zb, rqb, netf, cat384, NP);

    // disparity head
    conv(cat384, 384, w_f1h, fh1_b, t256, 256, 0, nullptr, 0, 0, 3, 3, 1, 4, 256, 16, 1, 1.f);
    conv(t256, 256, w_f2h, fh2_b, nullptr, 0, 0, delta, 1, 0, 3, 3, 1, 8, 1, 2, 0, 1.f);
    // mask head
    conv(cat384, 384, w_m1, mh1_b, t256, 256, 0, nullptr, 0, 0, 3, 3, 1, 4, 256, 16, 1, 1.f);
    conv(t256, 256, w_m2, mh2_b, nullptr, 0, 0, maskb, 576, 0, 1, 1, 0, 8, 576, 36, 0, 0.25f);

    k_coords_upd<<<grid(NP), TB, 0, stream>>>(coords, delta, disp2, W, NP);
    float* oslab = (float*)d_out + (size_t)it * B * OH * OW;
    k_upsample<<<grid((long)B * OH * OW), TB, 0, stream>>>(maskb, disp2, oslab, H, W, B);
  }
}
